// GroundingAttention_27754078667082
// MI455X (gfx1250) — matrix operand order repaired and checked by reading, not compiled, not run on hardware
//
#include <hip/hip_runtime.h>
#include <hip/hip_bf16.h>

typedef __attribute__((ext_vector_type(16))) __bf16 v16bf;
typedef __attribute__((ext_vector_type(8)))  __bf16 v8bf;
typedef __attribute__((ext_vector_type(8)))  float  v8f;

#define C_DIM   768
#define HEADS   12
#define HDIM    64
#define B_SZ    4
#define N_SZ    2048
#define M_SZ    1024
#define ATT_SCALE 0.125f   // 1/sqrt(64)

__device__ __forceinline__ v16bf cat16(v8bf lo, v8bf hi) {
    return __builtin_shufflevector(lo, hi, 0,1,2,3,4,5,6,7,8,9,10,11,12,13,14,15);
}

// ---------------- elementwise convert / transpose / repack ----------------

__global__ void cvt_f32_bf16(const float* __restrict__ in, __bf16* __restrict__ out, int n) {
    int i = blockIdx.x * blockDim.x + threadIdx.x;
    if (i < n) out[i] = (__bf16)in[i];
}

// in: f32 [K x N] row-major, out: bf16 [N x K] row-major (transposed)
__global__ void transpose_cvt(const float* __restrict__ in, __bf16* __restrict__ out,
                              int K, int N, int total) {
    int i = blockIdx.x * blockDim.x + threadIdx.x;
    if (i >= total) return;
    int k = i % K;
    int n = i / K;
    out[i] = (__bf16)in[(size_t)k * N + n];
}

// kvb: bf16 [B*M x 2C]; V part is cols [C, 2C). Out: vT[(b,h,d,m)]
__global__ void repack_vT(const __bf16* __restrict__ kvb, __bf16* __restrict__ vT, int total) {
    int i = blockIdx.x * blockDim.x + threadIdx.x;
    if (i >= total) return;
    int m = i % M_SZ;  int t = i / M_SZ;
    int d = t % HDIM;  t /= HDIM;
    int h = t % HEADS; int b = t / HEADS;
    vT[i] = kvb[((size_t)(b * M_SZ + m)) * (2 * C_DIM) + C_DIM + h * HDIM + d];
}

// ---------------- generic WMMA bf16 GEMM: C(bf16)[MxN] = A[MxK] * BT[NxK]^T --------

__global__ __launch_bounds__(256)
void gemm_bf16_bf16(const __bf16* __restrict__ A, const __bf16* __restrict__ BT,
                    __bf16* __restrict__ Cb, int M, int N, int K,
                    int tiles_n, int total_waves) {
    int widx = threadIdx.x >> 5;
    int wid  = blockIdx.x * (blockDim.x >> 5) + widx;
    if (wid >= total_waves) return;
    int lane = threadIdx.x & 31;
    int half = lane >> 4, lr = lane & 15;
    int tm = wid / tiles_n, tn = wid % tiles_n;
    int m0 = tm * 32, n0 = tn * 64;

    v8f acc[2][4] = {};
    for (int k0 = 0; k0 < K; k0 += 32) {
        v16bf a[2], b[4];
#pragma unroll
        for (int mt = 0; mt < 2; ++mt) {
            const __bf16* p = A + (size_t)(m0 + mt * 16 + lr) * K + k0 + 16 * half;
            a[mt] = *(const v16bf*)p;
        }
#pragma unroll
        for (int nt = 0; nt < 4; ++nt) {
            const __bf16* p = BT + (size_t)(n0 + nt * 16 + lr) * K + k0 + 16 * half;
            b[nt] = *(const v16bf*)p;
        }
#pragma unroll
        for (int mt = 0; mt < 2; ++mt)
#pragma unroll
            for (int nt = 0; nt < 4; ++nt)
                acc[mt][nt] = __builtin_amdgcn_wmma_f32_16x16x32_bf16(
                    false, a[mt], false, b[nt], (short)0, acc[mt][nt], false, false);
    }
#pragma unroll
    for (int mt = 0; mt < 2; ++mt)
#pragma unroll
        for (int nt = 0; nt < 4; ++nt)
#pragma unroll
            for (int r = 0; r < 8; ++r) {
                int row = m0 + mt * 16 + r + 8 * half;
                int col = n0 + nt * 16 + lr;
                Cb[(size_t)row * N + col] = (__bf16)acc[mt][nt][r];
            }
}

// ---------------- final WMMA GEMM with bias, f32 out ----------------

__global__ __launch_bounds__(256)
void gemm_bf16_f32_bias(const __bf16* __restrict__ A, const __bf16* __restrict__ BT,
                        const float* __restrict__ bias, float* __restrict__ Cf,
                        int M, int N, int K, int tiles_n, int total_waves) {
    int widx = threadIdx.x >> 5;
    int wid  = blockIdx.x * (blockDim.x >> 5) + widx;
    if (wid >= total_waves) return;
    int lane = threadIdx.x & 31;
    int half = lane >> 4, lr = lane & 15;
    int tm = wid / tiles_n, tn = wid % tiles_n;
    int m0 = tm * 32, n0 = tn * 64;

    v8f acc[2][4] = {};
    for (int k0 = 0; k0 < K; k0 += 32) {
        v16bf a[2], b[4];
#pragma unroll
        for (int mt = 0; mt < 2; ++mt) {
            const __bf16* p = A + (size_t)(m0 + mt * 16 + lr) * K + k0 + 16 * half;
            a[mt] = *(const v16bf*)p;
        }
#pragma unroll
        for (int nt = 0; nt < 4; ++nt) {
            const __bf16* p = BT + (size_t)(n0 + nt * 16 + lr) * K + k0 + 16 * half;
            b[nt] = *(const v16bf*)p;
        }
#pragma unroll
        for (int mt = 0; mt < 2; ++mt)
#pragma unroll
            for (int nt = 0; nt < 4; ++nt)
                acc[mt][nt] = __builtin_amdgcn_wmma_f32_16x16x32_bf16(
                    false, a[mt], false, b[nt], (short)0, acc[mt][nt], false, false);
    }
#pragma unroll
    for (int mt = 0; mt < 2; ++mt)
#pragma unroll
        for (int nt = 0; nt < 4; ++nt)
#pragma unroll
            for (int r = 0; r < 8; ++r) {
                int row = m0 + mt * 16 + r + 8 * half;
                int col = n0 + nt * 16 + lr;
                Cf[(size_t)row * N + col] = acc[mt][nt][r] + bias[col];
            }
}

// ---------------- flash attention: one wave per (b, h, 16 q-rows) ----------------

__global__ __launch_bounds__(256)
void attn_kernel(const __bf16* __restrict__ qb, const __bf16* __restrict__ kvb,
                 const __bf16* __restrict__ vT, __bf16* __restrict__ attn,
                 int total_waves) {
    __shared__ __align__(32) __bf16 pbuf[8 * 16 * 32];   // per-wave 16x32 bf16 P tile
    int widx = threadIdx.x >> 5;
    int wid  = blockIdx.x * 8 + widx;
    if (wid >= total_waves) return;
    int lane = threadIdx.x & 31, half = lane >> 4, lr = lane & 15;

    int tq = wid % (N_SZ / 16);
    int t  = wid / (N_SZ / 16);
    int h  = t % HEADS;
    int b  = t / HEADS;
    int n0 = tq * 16;

    // Q fragment: A-layout, rows n0..n0+15, head-dim chunks {0..31},{32..63}
    const __bf16* qrow = qb + ((size_t)(b * N_SZ + n0 + lr)) * C_DIM + h * HDIM;
    v16bf qf[2];
#pragma unroll
    for (int kc = 0; kc < 2; ++kc) {
        const __bf16* p = qrow + kc * 32 + 16 * half;
        qf[kc] = *(const v16bf*)p;
    }

    const __bf16* kbase = kvb + (size_t)b * M_SZ * (2 * C_DIM) + h * HDIM;   // K rows contiguous in d
    const __bf16* vbase = vT + ((size_t)(b * HEADS + h)) * HDIM * M_SZ;      // V^T rows contiguous in m

    v8f o[4] = {};
    float mr[8], lsum[8];
#pragma unroll
    for (int r = 0; r < 8; ++r) { mr[r] = -__builtin_inff(); lsum[r] = 0.0f; }

    __bf16* pw = pbuf + widx * 512;

    for (int m0 = 0; m0 < M_SZ; m0 += 32) {
        // ---- S = Q * K^T  (16 x 32 raw scores) ----
        v8f s[2] = {};
#pragma unroll
        for (int j = 0; j < 2; ++j) {
            const __bf16* krow = kbase + (size_t)(m0 + j * 16 + lr) * (2 * C_DIM);
#pragma unroll
            for (int kc = 0; kc < 2; ++kc) {
                v16bf kf = *(const v16bf*)(krow + kc * 32 + 16 * half);
                s[j] = __builtin_amdgcn_wmma_f32_16x16x32_bf16(
                    false, qf[kc], false, kf, (short)0, s[j], false, false);
            }
        }
        // ---- online softmax (scale folded into exponent) ----
#pragma unroll
        for (int r = 0; r < 8; ++r) {
            float v = fmaxf(s[0][r], s[1][r]);
            v = fmaxf(v, __shfl_xor(v, 1));
            v = fmaxf(v, __shfl_xor(v, 2));
            v = fmaxf(v, __shfl_xor(v, 4));
            v = fmaxf(v, __shfl_xor(v, 8));
            float mnew  = fmaxf(mr[r], v);
            float alpha = __expf(ATT_SCALE * (mr[r] - mnew));
            float p0 = __expf(ATT_SCALE * (s[0][r] - mnew));
            float p1 = __expf(ATT_SCALE * (s[1][r] - mnew));
            float ps = p0 + p1;
            ps += __shfl_xor(ps, 1);
            ps += __shfl_xor(ps, 2);
            ps += __shfl_xor(ps, 4);
            ps += __shfl_xor(ps, 8);
            lsum[r] = lsum[r] * alpha + ps;
            mr[r] = mnew;
#pragma unroll
            for (int dt = 0; dt < 4; ++dt) o[dt][r] *= alpha;
            // D-layout -> row-major bf16 P in LDS
            pw[(r + 8 * half) * 32 + lr]      = (__bf16)p0;
            pw[(r + 8 * half) * 32 + 16 + lr] = (__bf16)p1;
        }
        asm volatile("s_wait_dscnt 0x0" ::: "memory");
        // reload P in A-fragment layout (16x32)
        const __bf16* pp = pw + lr * 32 + 16 * half;
        v16bf pf = *(const v16bf*)pp;
        // ---- O += P * V  (four 16x16 d-tiles) ----
#pragma unroll
        for (int dt = 0; dt < 4; ++dt) {
            v16bf vf = *(const v16bf*)(vbase + (size_t)(dt * 16 + lr) * M_SZ + m0 + 16 * half);
            o[dt] = __builtin_amdgcn_wmma_f32_16x16x32_bf16(
                false, pf, false, vf, (short)0, o[dt], false, false);
        }
    }
    // ---- epilogue: normalize and write attn_out (bf16, [B*N x C]) ----
#pragma unroll
    for (int r = 0; r < 8; ++r) {
        float inv = 1.0f / lsum[r];
        int row = n0 + r + 8 * half;
        __bf16* orow = attn + ((size_t)(b * N_SZ + row)) * C_DIM + h * HDIM;
#pragma unroll
        for (int dt = 0; dt < 4; ++dt)
            orow[dt * 16 + lr] = (__bf16)(o[dt][r] * inv);
    }
}

// ---------------- host-side orchestration ----------------

extern "C" void kernel_launch(void* const* d_in, const int* in_sizes, int n_in,
                              void* d_out, int out_size, void* d_ws, size_t ws_size,
                              hipStream_t stream) {
    (void)in_sizes; (void)n_in; (void)out_size; (void)ws_size;
    const float* x      = (const float*)d_in[0];
    const float* r      = (const float*)d_in[1];
    const float* W_q    = (const float*)d_in[2];
    const float* W_kv   = (const float*)d_in[3];
    const float* W_proj = (const float*)d_in[4];
    const float* b_proj = (const float*)d_in[5];
    float* out = (float*)d_out;

    char*  ws  = (char*)d_ws;
    size_t off = 0;
    auto alloc = [&](size_t bytes) -> void* {
        void* p = ws + off;
        off = (off + bytes + 255) & ~(size_t)255;
        return p;
    };
    const size_t BN = (size_t)B_SZ * N_SZ;   // 8192
    const size_t BM = (size_t)B_SZ * M_SZ;   // 4096
    __bf16* xb   = (__bf16*)alloc(BN * C_DIM * 2);
    __bf16* rb   = (__bf16*)alloc(BM * C_DIM * 2);
    __bf16* wqT  = (__bf16*)alloc((size_t)C_DIM * C_DIM * 2);
    __bf16* wkvT = (__bf16*)alloc((size_t)2 * C_DIM * C_DIM * 2);
    __bf16* wpT  = (__bf16*)alloc((size_t)C_DIM * C_DIM * 2);
    __bf16* qb   = (__bf16*)alloc(BN * C_DIM * 2);
    __bf16* kvb  = (__bf16*)alloc(BM * 2 * C_DIM * 2);
    __bf16* vTb  = (__bf16*)alloc((size_t)B_SZ * HEADS * HDIM * M_SZ * 2);
    __bf16* attn = (__bf16*)alloc(BN * C_DIM * 2);

    // 1. convert activations
    {
        int n = (int)(BN * C_DIM);
        cvt_f32_bf16<<<(n + 255) / 256, 256, 0, stream>>>(x, xb, n);
        n = (int)(BM * C_DIM);
        cvt_f32_bf16<<<(n + 255) / 256, 256, 0, stream>>>(r, rb, n);
    }
    // 2. transpose+convert weights to [N x K] bf16
    {
        int tot = C_DIM * C_DIM;
        transpose_cvt<<<(tot + 255) / 256, 256, 0, stream>>>(W_q, wqT, C_DIM, C_DIM, tot);
        transpose_cvt<<<(tot + 255) / 256, 256, 0, stream>>>(W_proj, wpT, C_DIM, C_DIM, tot);
        int tot2 = 2 * C_DIM * C_DIM;
        transpose_cvt<<<(tot2 + 255) / 256, 256, 0, stream>>>(W_kv, wkvT, C_DIM, 2 * C_DIM, tot2);
    }
    // 3. Q = xb * W_q   (M=8192, N=768, K=768)
    {
        int tiles_n = C_DIM / 64, waves = (int)(BN / 32) * tiles_n;
        gemm_bf16_bf16<<<(waves + 7) / 8, 256, 0, stream>>>(xb, wqT, qb,
            (int)BN, C_DIM, C_DIM, tiles_n, waves);
    }
    // 4. KV = rb * W_kv (M=4096, N=1536, K=768)
    {
        int tiles_n = (2 * C_DIM) / 64, waves = (int)(BM / 32) * tiles_n;
        gemm_bf16_bf16<<<(waves + 7) / 8, 256, 0, stream>>>(rb, wkvT, kvb,
            (int)BM, 2 * C_DIM, C_DIM, tiles_n, waves);
    }
    // 5. repack V -> V^T (b,h,d,m)
    {
        int tot = B_SZ * HEADS * HDIM * M_SZ;
        repack_vT<<<(tot + 255) / 256, 256, 0, stream>>>(kvb, vTb, tot);
    }
    // 6. flash attention
    {
        int waves = B_SZ * HEADS * (N_SZ / 16);   // 6144
        attn_kernel<<<(waves + 7) / 8, 256, 0, stream>>>(qb, kvb, vTb, attn, waves);
    }
    // 7. out = attn * W_proj + b_proj (f32)
    {
        int tiles_n = C_DIM / 64, waves = (int)(BN / 32) * tiles_n;
        gemm_bf16_f32_bias<<<(waves + 7) / 8, 256, 0, stream>>>(attn, wpT, b_proj, out,
            (int)BN, C_DIM, C_DIM, tiles_n, waves);
    }
}
